// GraphConvolution_71605694759080
// MI455X (gfx1250) — matrix operand order repaired and checked by reading, not compiled, not run on hardware
//
#include <hip/hip_runtime.h>
#include <hip/hip_bf16.h>

#define N_NODES 50000
#define N_EDGES 400000
#define D_INN   512
#define D_OUTN  512

typedef __attribute__((ext_vector_type(16))) __bf16 v16bf;
typedef __attribute__((ext_vector_type(8)))  float  v8f;

union FragB16 {
  uint4 q[2];
  v16bf v;
};

__device__ __forceinline__ unsigned short f32_to_bf16(float f) {
  unsigned int u = __builtin_bit_cast(unsigned int, f);
  u += 0x7fffu + ((u >> 16) & 1u);          // round-to-nearest-even
  return (unsigned short)(u >> 16);
}

__device__ __forceinline__ float bf16_to_f32(unsigned short h) {
  unsigned int u = ((unsigned int)h) << 16;
  return __builtin_bit_cast(float, u);
}

// ---------------------------------------------------------------------------
// Split x (row-major, M x K) into bf16 hi/lo arrays (same layout).
// ---------------------------------------------------------------------------
__global__ void pack_x_kernel(const float* __restrict__ x,
                              unsigned short* __restrict__ xh,
                              unsigned short* __restrict__ xl) {
  int i = blockIdx.x * blockDim.x + threadIdx.x;
  const int n = N_NODES * D_INN;
  const int stride = gridDim.x * blockDim.x;
  for (; i < n; i += stride) {
    float f = x[i];
    unsigned short h = f32_to_bf16(f);
    xh[i] = h;
    xl[i] = f32_to_bf16(f - bf16_to_f32(h));
  }
}

// ---------------------------------------------------------------------------
// Split W (row-major K x N) into bf16 hi/lo, TRANSPOSED to N-major (wt[n][k])
// so each lane's B-fragment is two contiguous 8-element runs along K.
// ---------------------------------------------------------------------------
__global__ void pack_w_kernel(const float* __restrict__ w,
                              unsigned short* __restrict__ wht,
                              unsigned short* __restrict__ wlt) {
  int i = blockIdx.x * blockDim.x + threadIdx.x;
  const int n = D_INN * D_OUTN;
  if (i >= n) return;
  int k = i / D_OUTN;
  int c = i % D_OUTN;
  float f = w[i];
  unsigned short h = f32_to_bf16(f);
  wht[c * D_INN + k] = h;
  wlt[c * D_INN + k] = f32_to_bf16(f - bf16_to_f32(h));
}

// ---------------------------------------------------------------------------
// GEMM: support = x @ W in near-fp32 precision via 3-way bf16 split WMMA.
// Block = 256 threads = 8 waves. Block covers 32 rows x all 512 cols.
// Wave w covers the 32x64 tile at n0 = w*64: two 16-row accumulator sets so
// each B fragment feeds 6 WMMAs (24 WMMAs / 24 b128 loads per k-step).
// A-fragment addresses are identical across the 8 waves (WGP$ broadcast).
// grid.x = ceil(50000/32) = 1563; last block clamps its second row tile.
// ---------------------------------------------------------------------------
__global__ void gemm_kernel(const unsigned short* __restrict__ xh,
                            const unsigned short* __restrict__ xl,
                            const unsigned short* __restrict__ wht,
                            const unsigned short* __restrict__ wlt,
                            float* __restrict__ support) {
  const int wv   = threadIdx.x >> 5;      // wave in block: 0..7
  const int lane = threadIdx.x & 31;
  const int half = lane >> 4;             // 0 | 1
  const int lidx = lane & 15;
  const int m0 = blockIdx.x * 32;
  const int n0 = wv * 64;

  v8f acc[2][4] = {};

  // Row indices for this lane's two A tiles; clamp keeps loads in-bounds
  // without branches (WMMA requires full EXEC inside the k-loop).
  const int r0 = m0 + lidx;                                   // < 50000 always
  int r1 = m0 + 16 + lidx;
  if (r1 > N_NODES - 1) r1 = N_NODES - 1;                     // last block only

  const unsigned short* a0h = xh + (size_t)r0 * D_INN;
  const unsigned short* a0l = xl + (size_t)r0 * D_INN;
  const unsigned short* a1h = xh + (size_t)r1 * D_INN;
  const unsigned short* a1l = xl + (size_t)r1 * D_INN;

  for (int k = 0; k < D_INN; k += 32) {
    FragB16 ah[2], al[2];
    // A 16x32 bf16 layout: lane half 0 -> K[k..k+7], K[k+16..k+23]
    //                      lane half 1 -> K[k+8..k+15], K[k+24..k+31]
    ah[0].q[0] = *(const uint4*)(a0h + k + half * 8);
    ah[0].q[1] = *(const uint4*)(a0h + k + 16 + half * 8);
    al[0].q[0] = *(const uint4*)(a0l + k + half * 8);
    al[0].q[1] = *(const uint4*)(a0l + k + 16 + half * 8);
    ah[1].q[0] = *(const uint4*)(a1h + k + half * 8);
    ah[1].q[1] = *(const uint4*)(a1h + k + 16 + half * 8);
    al[1].q[0] = *(const uint4*)(a1l + k + half * 8);
    al[1].q[1] = *(const uint4*)(a1l + k + 16 + half * 8);

#pragma unroll
    for (int j = 0; j < 4; ++j) {
      const int n = n0 + j * 16 + lidx;
      // B 32x16 bf16 layout: lane (N=lidx), K[k+half*8..+7], K[k+16+half*8..+7]
      const unsigned short* bch = wht + (size_t)n * D_INN + k + half * 8;
      const unsigned short* bcl = wlt + (size_t)n * D_INN + k + half * 8;
      FragB16 bh, bl;
      bh.q[0] = *(const uint4*)(bch);
      bh.q[1] = *(const uint4*)(bch + 16);
      bl.q[0] = *(const uint4*)(bcl);
      bl.q[1] = *(const uint4*)(bcl + 16);

#pragma unroll
      for (int t = 0; t < 2; ++t) {
        acc[t][j] = __builtin_amdgcn_wmma_f32_16x16x32_bf16(
            false, ah[t].v, false, bh.v, (short)0, acc[t][j], false, false);
        acc[t][j] = __builtin_amdgcn_wmma_f32_16x16x32_bf16(
            false, ah[t].v, false, bl.v, (short)0, acc[t][j], false, false);
        acc[t][j] = __builtin_amdgcn_wmma_f32_16x16x32_bf16(
            false, al[t].v, false, bh.v, (short)0, acc[t][j], false, false);
      }
    }
  }

  // C/D layout: element r of acc -> row m0 + t*16 + r + half*8, col n0+j*16+lidx
#pragma unroll
  for (int t = 0; t < 2; ++t) {
#pragma unroll
    for (int j = 0; j < 4; ++j) {
      const int n = n0 + j * 16 + lidx;
#pragma unroll
      for (int r = 0; r < 8; ++r) {
        const int mrow = m0 + t * 16 + r + half * 8;
        if (mrow < N_NODES)
          support[(size_t)mrow * D_OUTN + n] = acc[t][j][r];
      }
    }
  }
}

// ---------------------------------------------------------------------------
// CSR construction
// ---------------------------------------------------------------------------
__global__ void zero_counts_kernel(int* __restrict__ counts) {
  int i = blockIdx.x * blockDim.x + threadIdx.x;
  if (i < N_NODES) counts[i] = 0;
}

__global__ void count_kernel(const int* __restrict__ adj_rows,
                             int* __restrict__ counts) {
  int e = blockIdx.x * blockDim.x + threadIdx.x;
  if (e < N_EDGES) atomicAdd(&counts[adj_rows[e]], 1);
}

#define SCAN_T 256
#define SCAN_CHUNK ((N_NODES + SCAN_T - 1) / SCAN_T)   // 196

__global__ void scan_kernel(const int* __restrict__ counts,
                            int* __restrict__ row_start,
                            int* __restrict__ cursor) {
  __shared__ int partial[SCAN_T];
  const int t = threadIdx.x;
  const int beg = t * SCAN_CHUNK;
  const int end = (beg + SCAN_CHUNK < N_NODES) ? (beg + SCAN_CHUNK) : N_NODES;
  int s = 0;
  for (int i = beg; i < end; ++i) s += counts[i];
  partial[t] = s;
  __syncthreads();
  if (t == 0) {
    int acc = 0;
    for (int i = 0; i < SCAN_T; ++i) { int v = partial[i]; partial[i] = acc; acc += v; }
  }
  __syncthreads();
  int off = partial[t];
  for (int i = beg; i < end; ++i) {
    row_start[i] = off;
    cursor[i]    = off;
    off += counts[i];
  }
  if (t == SCAN_T - 1) row_start[N_NODES] = off;   // == N_EDGES
}

__global__ void scatter_kernel(const int* __restrict__ adj_rows,
                               int* __restrict__ cursor,
                               int* __restrict__ edge_perm) {
  int e = blockIdx.x * blockDim.x + threadIdx.x;
  if (e < N_EDGES) {
    int pos = atomicAdd(&cursor[adj_rows[e]], 1);
    edge_perm[pos] = e;
  }
}

// ---------------------------------------------------------------------------
// SpMM: one wave32 per output row. Each lane holds 16 output floats (4x float4)
// in registers; atomics fully eliminated. support (102 MB) is L2-resident;
// the next edge's 2 KB support row is prefetched (32 lanes x 64 B spacing)
// to overlap L2 latency with the current edge's FMAs.
// ---------------------------------------------------------------------------
__global__ void spmm_kernel(const int* __restrict__ row_start,
                            const int* __restrict__ edge_perm,
                            const int* __restrict__ adj_cols,
                            const float* __restrict__ adj_vals,
                            const float* __restrict__ support,
                            const float* __restrict__ bias,
                            float* __restrict__ out) {
  const int row  = __builtin_amdgcn_readfirstlane(
      (int)((blockIdx.x * blockDim.x + threadIdx.x) >> 5));
  const int lane = threadIdx.x & 31;
  if (row >= N_NODES) return;

  const int s = row_start[row];
  const int e = row_start[row + 1];

  float4 acc[4];
#pragma unroll
  for (int t = 0; t < 4; ++t) acc[t] = make_float4(0.f, 0.f, 0.f, 0.f);

  for (int i = s; i < e; ++i) {
    const int   eid = edge_perm[i];
    const int   c   = adj_cols[eid];
    const float v   = adj_vals[eid];

    if (i + 1 < e) {
      const int nc = adj_cols[edge_perm[i + 1]];
      __builtin_prefetch(support + (size_t)nc * D_OUTN + lane * 16, 0, 1);
    }

    const float4* sp = (const float4*)(support + (size_t)c * D_OUTN);
#pragma unroll
    for (int t = 0; t < 4; ++t) {
      float4 d = sp[t * 32 + lane];
      acc[t].x += v * d.x;
      acc[t].y += v * d.y;
      acc[t].z += v * d.z;
      acc[t].w += v * d.w;
    }
  }

  const float4* b4 = (const float4*)bias;
  float4* o4 = (float4*)(out + (size_t)row * D_OUTN);
#pragma unroll
  for (int t = 0; t < 4; ++t) {
    float4 bb = b4[t * 32 + lane];
    float4 r  = acc[t];
    r.x += bb.x; r.y += bb.y; r.z += bb.z; r.w += bb.w;
    o4[t * 32 + lane] = r;
  }
}

// ---------------------------------------------------------------------------
// Launcher
// ---------------------------------------------------------------------------
extern "C" void kernel_launch(void* const* d_in, const int* in_sizes, int n_in,
                              void* d_out, int out_size, void* d_ws, size_t ws_size,
                              hipStream_t stream) {
  const float* x        = (const float*)d_in[0];
  const int*   adj_rows = (const int*)  d_in[1];
  const int*   adj_cols = (const int*)  d_in[2];
  const float* adj_vals = (const float*)d_in[3];
  const float* weight   = (const float*)d_in[4];
  const float* bias     = (const float*)d_in[5];
  float*       out      = (float*)d_out;

  // Workspace carve-up (all chunks 16B-aligned)
  char* ws = (char*)d_ws;
  float* support = (float*)ws;               ws += (size_t)N_NODES * D_OUTN * 4;  // 102.4 MB
  unsigned short* xh  = (unsigned short*)ws; ws += (size_t)N_NODES * D_INN * 2;   //  51.2 MB
  unsigned short* xl  = (unsigned short*)ws; ws += (size_t)N_NODES * D_INN * 2;   //  51.2 MB
  unsigned short* wht = (unsigned short*)ws; ws += (size_t)D_INN * D_OUTN * 2;    //   0.5 MB
  unsigned short* wlt = (unsigned short*)ws; ws += (size_t)D_INN * D_OUTN * 2;    //   0.5 MB
  int* counts    = (int*)ws;                 ws += ((size_t)N_NODES + 4) * 4;
  int* row_start = (int*)ws;                 ws += ((size_t)N_NODES + 4) * 4;
  int* cursor    = (int*)ws;                 ws += ((size_t)N_NODES + 4) * 4;
  int* edge_perm = (int*)ws;                 ws += (size_t)N_EDGES * 4;
  (void)ws_size; (void)in_sizes; (void)n_in; (void)out_size;

  // Phase 0: precision-split packing
  pack_x_kernel<<<2048, 256, 0, stream>>>(x, xh, xl);
  pack_w_kernel<<<(D_INN * D_OUTN + 255) / 256, 256, 0, stream>>>(weight, wht, wlt);

  // Phase 1: CSR build
  zero_counts_kernel<<<(N_NODES + 255) / 256, 256, 0, stream>>>(counts);
  count_kernel<<<(N_EDGES + 255) / 256, 256, 0, stream>>>(adj_rows, counts);
  scan_kernel<<<1, SCAN_T, 0, stream>>>(counts, row_start, cursor);
  scatter_kernel<<<(N_EDGES + 255) / 256, 256, 0, stream>>>(adj_rows, cursor, edge_perm);

  // Phase 2: WMMA GEMM  support = x @ W
  gemm_kernel<<<(N_NODES + 31) / 32, 256, 0, stream>>>(xh, xl, wht, wlt, support);

  // Phase 3: atomic-free SpMM + bias, one wave per row
  spmm_kernel<<<(N_NODES * 32 + 255) / 256, 256, 0, stream>>>(
      row_start, edge_perm, adj_cols, adj_vals, support, bias, out);
}
